// Model_21208548507911
// MI455X (gfx1250) — compile-verified, matrix operand order checked by reading, not run on hardware
//
#include <hip/hip_runtime.h>

// ---- model constants ----
#define R_ROWS   24           // grid rows
#define C_COLS   48           // grid cols
#define L_STEPS  71           // R + C - 1 diagonal steps
#define BATCH    64
#define MROWS    1536         // R * BATCH
#define H        256
#define SIXH     1536         // 6*H (gate width)
#define SEQ      1152
#define ENC      21
#define TDIM     4
#define NL       2
#define KPAD0    544          // 2H + 32 (25-feature input padded to 32)
#define KPAD1    1024         // 2H + 2H

typedef __attribute__((ext_vector_type(16))) __bf16 v16bf;
typedef __attribute__((ext_vector_type(8)))  float  v8f;

__device__ __forceinline__ unsigned short f2bf(float f) {
  unsigned int u = __float_as_uint(f);
  u += 0x7FFFu + ((u >> 16) & 1u);          // round-to-nearest-even
  return (unsigned short)(u >> 16);
}
__device__ __forceinline__ float sigmoidf_(float x) {
  return 1.0f / (1.0f + __expf(-x));
}

// Async global->LDS copy, 16B per lane, tracked by ASYNCcnt (CDNA5 TDM-lite path).
// LDS address = low 32 bits of the generic shared pointer (flat->LDS truncation).
__device__ __forceinline__ void async_b128(const unsigned short* g, unsigned short* l) {
  unsigned int lo = (unsigned int)(size_t)l;
  asm volatile("global_load_async_to_lds_b128 %0, %1, off"
               :: "v"(lo), "v"(g) : "memory");
}
#define S_WAIT_ASYNCCNT(n) asm volatile("s_wait_asynccnt " #n ::: "memory")

// ---------------------------------------------------------------------------
// Build layer-0 parallelogram input: a0[l][m][f] (bf16), m = r*64+b, f<32.
// a0 = (r <= l < r+C && f<25) ? [x_enc - seq_last | x_mark_enc] : 0
// ---------------------------------------------------------------------------
__global__ __launch_bounds__(256) void build_a0_kernel(
    const float* __restrict__ x_enc, const float* __restrict__ x_mark,
    unsigned short* __restrict__ a0)
{
  int idx = blockIdx.x * 256 + threadIdx.x;
  if (idx >= L_STEPS * MROWS * 32) return;
  int f  = idx & 31;
  int lm = idx >> 5;
  int m  = lm % MROWS;
  int l  = lm / MROWS;
  int r  = m >> 6;
  int b  = m & 63;
  int c  = l - r;
  float v = 0.0f;
  if (c >= 0 && c < C_COLS && f < 25) {
    int t = r * C_COLS + c;
    if (f < ENC)
      v = x_enc[((size_t)b * SEQ + t) * ENC + f]
        - x_enc[((size_t)b * SEQ + (SEQ - 1)) * ENC + f];
    else
      v = x_mark[((size_t)b * SEQ + t) * TDIM + (f - ENC)];
  }
  a0[idx] = f2bf(v);
}

// Convert a (rows x kin) f32 weight matrix to (rows x kpad) bf16, zero padded.
__global__ __launch_bounds__(256) void convert_w_kernel(
    const float* __restrict__ W, unsigned short* __restrict__ Wb,
    int kin, int kpad, int total)
{
  int idx = blockIdx.x * 256 + threadIdx.x;
  if (idx >= total) return;
  int k = idx % kpad;
  int n = idx / kpad;
  Wb[idx] = f2bf(k < kin ? W[(size_t)n * kin + k] : 0.0f);
}

__global__ __launch_bounds__(256) void zero_kernel(unsigned int* __restrict__ p, int n) {
  int i = blockIdx.x * 256 + threadIdx.x;
  if (i < n) p[i] = 0u;
}

// ---------------------------------------------------------------------------
// gate = X @ W^T, X = [h_row | h_col | a_s]  (M=1536, N=1536, K=Kpad)
// bf16 inputs, f32 accumulate via v_wmma_f32_16x16x32_bf16.
// WG tile 64(M) x 128(N), 8 waves each 32x32. K staged in 32-chunks through
// double-buffered LDS filled by global_load_async_to_lds_b128 (ASYNCcnt):
// the async engine fills chunk k+1 while WMMAs consume chunk k.
// LDS row stride 40 bf16 -> conflict-free 16-row fragment reads.
// ---------------------------------------------------------------------------
#define LDSW 40
__global__ __launch_bounds__(256) void wmma_gate_gemm(
    const unsigned short* __restrict__ Xhid,   // 1536 x 512 bf16 (h_row|h_col)
    const unsigned short* __restrict__ Aext,   // 1536 x Fext bf16 (step input)
    int Fext,
    const unsigned short* __restrict__ Wb,     // 1536 x Kpad bf16
    int Kpad,
    float* __restrict__ gate)                  // 1536 x 1536 f32
{
  __shared__ unsigned short lA[2][64 * LDSW];
  __shared__ unsigned short lB[2][128 * LDSW];
  const int tid  = threadIdx.x;
  const int lane = tid & 31;
  const int wave = tid >> 5;
  const int wm   = wave >> 2;          // 0..1  -> M sub-tile of 32
  const int wn   = wave & 3;           // 0..3  -> N sub-tile of 32
  const int n0   = blockIdx.x * 128;
  const int m0   = blockIdx.y * 64;

  v8f acc[2][2];
  for (int i = 0; i < 2; ++i)
    for (int j = 0; j < 2; ++j)
      for (int e = 0; e < 8; ++e) acc[i][j][e] = 0.0f;

  const int lrow = tid >> 2;           // 0..63
  const int lcol = (tid & 3) * 8;      // 0,8,16,24 (bf16)
  const int mr   = lane & 15;
  const int hi   = lane >> 4;          // K-half select per ISA fragment layout

  // issue one K-chunk (3 async b128 per thread: 1 A-tile + 2 B-tile)
  auto issue = [&](int k0, int buf) {
    const unsigned short* asrc;
    int lda, kk;
    if (k0 < 512) { asrc = Xhid; lda = 512;  kk = k0; }
    else          { asrc = Aext; lda = Fext; kk = k0 - 512; }
    async_b128(asrc + (size_t)(m0 + lrow) * lda + kk + lcol,
               &lA[buf][lrow * LDSW + lcol]);
    async_b128(Wb + (size_t)(n0 + lrow) * Kpad + k0 + lcol,
               &lB[buf][lrow * LDSW + lcol]);
    async_b128(Wb + (size_t)(n0 + lrow + 64) * Kpad + k0 + lcol,
               &lB[buf][(lrow + 64) * LDSW + lcol]);
  };

  const int nk = Kpad >> 5;
  issue(0, 0);
  for (int c = 0; c < nk; ++c) {
    const int buf = c & 1;
    if (c + 1 < nk) {
      issue((c + 1) * 32, buf ^ 1);   // prefetch next chunk into other buffer
      S_WAIT_ASYNCCNT(3);             // chunk c landed; next trio may fly
    } else {
      S_WAIT_ASYNCCNT(0);
    }
    __syncthreads();                  // whole tile visible to all waves

    // Fragment loads per 16-bit WMMA layout:
    //  lanes 0-15  : K = {0..7, 16..23};  lanes 16-31 : K = {8..15, 24..31}
    union Frag { v16bf v; uint4 q[2]; };
    Frag af[2], bfr[2];
    #pragma unroll
    for (int i = 0; i < 2; ++i) {
      int row = wm * 32 + i * 16 + mr;
      af[i].q[0] = *(const uint4*)(&lA[buf][row * LDSW + hi * 8]);
      af[i].q[1] = *(const uint4*)(&lA[buf][row * LDSW + 16 + hi * 8]);
    }
    #pragma unroll
    for (int j = 0; j < 2; ++j) {
      int row = wn * 32 + j * 16 + mr;
      bfr[j].q[0] = *(const uint4*)(&lB[buf][row * LDSW + hi * 8]);
      bfr[j].q[1] = *(const uint4*)(&lB[buf][row * LDSW + 16 + hi * 8]);
    }
    #pragma unroll
    for (int i = 0; i < 2; ++i)
      #pragma unroll
      for (int j = 0; j < 2; ++j)
        acc[i][j] = __builtin_amdgcn_wmma_f32_16x16x32_bf16(
            false, af[i].v, false, bfr[j].v, (short)0, acc[i][j], false, false);
    __syncthreads();                  // done reading buf before it is refilled
  }

  // D layout: lane -> N = lane&15, element e -> M = e + 8*(lane>>4)
  const int dn = lane & 15;
  const int dm = (lane >> 4) * 8;
  #pragma unroll
  for (int i = 0; i < 2; ++i)
    #pragma unroll
    for (int j = 0; j < 2; ++j) {
      int gm = m0 + wm * 32 + i * 16 + dm;
      int gn = n0 + wn * 32 + j * 16 + dn;
      #pragma unroll
      for (int e = 0; e < 8; ++e)
        gate[(size_t)(gm + e) * SIXH + gn] = acc[i][j][e];
    }
}

// ---------------------------------------------------------------------------
// Per-step gating: masked bias, sigmoid/tanh GRU-like update, roll(h_col, B),
// write bf16 states for next step's GEMM, layer outputs, and extractions.
// ---------------------------------------------------------------------------
__global__ __launch_bounds__(256) void gating_step(
    const float* __restrict__ gate,      // 1536 x 1536
    const float* __restrict__ bias,      // 1536 (this layer)
    float* __restrict__ h_row,           // 1536 x 256 (in/out)
    const float* __restrict__ h_col_in,  // 1536 x 256 (rolled carry)
    float* __restrict__ h_col_out,       // 1536 x 256 (rolled for next step)
    unsigned short* __restrict__ Xhid,   // 1536 x 512 bf16
    unsigned short* __restrict__ outb,   // step slice 1536 x 512 bf16, or null
    float* __restrict__ hid_row,         // (B, NL, R, H)
    float* __restrict__ hid_col,         // (B, NL, C, H)
    int s, int layer)
{
  int idx = blockIdx.x * 256 + threadIdx.x;   // m*256 + j
  int m = idx >> 8;
  int j = idx & 255;
  const float* g = gate + (size_t)m * SIXH;
  float g0 = g[j],        g1 = g[256 + j],  g2 = g[512 + j];
  float g3 = g[768 + j],  g4 = g[1024 + j], g5 = g[1280 + j];
  int r = m >> 6;
  if (r <= s && s < R_ROWS) {
    g0 += bias[j];        g1 += bias[256 + j];  g2 += bias[512 + j];
    g3 += bias[768 + j];  g4 += bias[1024 + j]; g5 += bias[1280 + j];
  }
  float hr = h_row[idx];
  float hc = h_col_in[idx];
  float ur = sigmoidf_(g0), outr = sigmoidf_(g1);
  float uc = sigmoidf_(g2), outc = sigmoidf_(g3);
  float ir = tanhf(g4),     ic   = tanhf(g5);
  float hrn = tanhf((1.0f - ur) * hr + ur * ir) * outr;
  float hcn = tanhf((1.0f - uc) * hc + uc * ic) * outc;
  h_row[idx] = hrn;
  int mro = m + BATCH; if (mro >= MROWS) mro -= MROWS;   // roll by B
  h_col_out[(size_t)mro * 256 + j] = hcn;
  Xhid[(size_t)m   * 512 + j]       = f2bf(hrn);
  Xhid[(size_t)mro * 512 + 256 + j] = f2bf(hcn);
  if (outb) {  // layer output = concat(h_row, h_col_new), un-rolled
    outb[(size_t)m * 512 + j]       = f2bf(hrn);
    outb[(size_t)m * 512 + 256 + j] = f2bf(hcn);
  }
  int b = m & 63;
  if (s == (C_COLS - 1) + r)      // row r finishes at step C-1+r
    hid_row[((size_t)(b * NL + layer) * R_ROWS + r) * 256 + j] = hrn;
  if (r == R_ROWS - 1 && s >= R_ROWS - 1)
    hid_col[((size_t)(b * NL + layer) * C_COLS + (s - (R_ROWS - 1))) * 256 + j] = hcn;
}

// ---------------------------------------------------------------------------
// fc1: (B*C, 1024) @ fc1_W^T + b  -> last0 (B, C, 512)
// input k = l*512 + {0..255: hid_row[b][l][R-1], 256..511: hid_col[b][l][c]}
// ---------------------------------------------------------------------------
__global__ __launch_bounds__(256) void fc1_kernel(
    const float* __restrict__ hid_row, const float* __restrict__ hid_col,
    const float* __restrict__ W, const float* __restrict__ bvec,
    float* __restrict__ last0)
{
  int idx = blockIdx.x * 256 + threadIdx.x;   // (b*48+c)*512 + o
  int o  = idx & 511;
  int bc = idx >> 9;
  int c  = bc % C_COLS;
  int b  = bc / C_COLS;
  const float* wr = W + (size_t)o * 1024;
  float acc = bvec[o];
  #pragma unroll
  for (int l = 0; l < NL; ++l) {
    const float* hrp = hid_row + ((size_t)(b * NL + l) * R_ROWS + (R_ROWS - 1)) * 256;
    const float* hcp = hid_col + ((size_t)(b * NL + l) * C_COLS + c) * 256;
    for (int d = 0; d < 256; ++d) acc += hrp[d] * wr[l * 512 + d];
    for (int d = 0; d < 256; ++d) acc += hcp[d] * wr[l * 512 + 256 + d];
  }
  last0[idx] = acc;
}

// ---------------------------------------------------------------------------
// final: add temporal embedding, fc2, seq_last -> out (B, 96, 21)
// ---------------------------------------------------------------------------
__global__ __launch_bounds__(256) void final_kernel(
    const float* __restrict__ last0, const float* __restrict__ x_mark_dec,
    const float* __restrict__ temb_W, const float* __restrict__ fc2_W,
    const float* __restrict__ fc2_b, const float* __restrict__ x_enc,
    float* __restrict__ out)
{
  int idx = blockIdx.x * 256 + threadIdx.x;
  if (idx >= BATCH * 96 * 21) return;
  int o = idx % 21;
  int p = (idx / 21) % 96;
  int b = idx / (21 * 96);
  int dr = p / C_COLS, c = p % C_COLS;
  const float* u  = last0 + ((size_t)(b * C_COLS + c)) * 512 + dr * 256;
  const float* xm = x_mark_dec + ((size_t)b * 96 + p) * TDIM;
  float t0 = xm[0], t1 = xm[1], t2 = xm[2], t3 = xm[3];
  const float* w2 = fc2_W + (size_t)o * 256;
  float acc = fc2_b[o];
  for (int d = 0; d < 256; ++d) {
    const float* tw = temb_W + (size_t)d * TDIM;
    float dec = t0 * tw[0] + t1 * tw[1] + t2 * tw[2] + t3 * tw[3];
    acc += (u[d] + dec) * w2[d];
  }
  acc += x_enc[((size_t)b * SEQ + (SEQ - 1)) * ENC + o];
  out[idx] = acc;
}

// ---------------------------------------------------------------------------
extern "C" void kernel_launch(void* const* d_in, const int* in_sizes, int n_in,
                              void* d_out, int out_size, void* d_ws, size_t ws_size,
                              hipStream_t stream) {
  const float* x_enc      = (const float*)d_in[0];
  const float* x_mark_enc = (const float*)d_in[1];
  const float* x_mark_dec = (const float*)d_in[3];   // d_in[2] (x_dec) unused
  const float* W_first    = (const float*)d_in[4];
  const float* W_other    = (const float*)d_in[5];
  const float* Bias       = (const float*)d_in[6];
  const float* temb_W     = (const float*)d_in[7];
  const float* fc1_W      = (const float*)d_in[8];
  const float* fc1_b      = (const float*)d_in[9];
  const float* fc2_W      = (const float*)d_in[10];
  const float* fc2_b      = (const float*)d_in[11];
  float* out = (float*)d_out;

  char* ws = (char*)d_ws;
  size_t off = 0;
  auto alloc = [&](size_t bytes) -> void* {
    void* p = ws + off;
    off = (off + bytes + 255) & ~(size_t)255;
    return p;
  };
  unsigned short* a0    = (unsigned short*)alloc((size_t)L_STEPS * MROWS * 32 * 2);
  unsigned short* out0  = (unsigned short*)alloc((size_t)L_STEPS * MROWS * 512 * 2);
  unsigned short* Wb0   = (unsigned short*)alloc((size_t)SIXH * KPAD0 * 2);
  unsigned short* Wb1   = (unsigned short*)alloc((size_t)SIXH * KPAD1 * 2);
  unsigned short* Xhid  = (unsigned short*)alloc((size_t)MROWS * 512 * 2);
  float* hrow  = (float*)alloc((size_t)MROWS * 256 * 4);
  float* hcolA = (float*)alloc((size_t)MROWS * 256 * 4);
  float* hcolB = (float*)alloc((size_t)MROWS * 256 * 4);
  float* gate  = (float*)alloc((size_t)MROWS * SIXH * 4);
  float* hid_row = (float*)alloc((size_t)BATCH * NL * R_ROWS * 256 * 4);
  float* hid_col = (float*)alloc((size_t)BATCH * NL * C_COLS * 256 * 4);
  float* last0   = (float*)alloc((size_t)BATCH * C_COLS * 512 * 4);

  // --- preprocess ---
  build_a0_kernel<<<(L_STEPS * MROWS * 32 + 255) / 256, 256, 0, stream>>>(
      x_enc, x_mark_enc, a0);
  convert_w_kernel<<<(SIXH * KPAD0 + 255) / 256, 256, 0, stream>>>(
      W_first, Wb0, 537, KPAD0, SIXH * KPAD0);
  convert_w_kernel<<<(SIXH * KPAD1 + 255) / 256, 256, 0, stream>>>(
      W_other, Wb1, KPAD1, KPAD1, SIXH * KPAD1);

  // --- recurrent encoder: 2 layers x 71 sequential steps ---
  for (int layer = 0; layer < NL; ++layer) {
    zero_kernel<<<1536, 256, 0, stream>>>((unsigned int*)hrow,  MROWS * 256);
    zero_kernel<<<1536, 256, 0, stream>>>((unsigned int*)hcolA, MROWS * 256);
    zero_kernel<<<1536, 256, 0, stream>>>((unsigned int*)Xhid,  MROWS * 512 / 2);
    const unsigned short* Wb = (layer == 0) ? Wb0 : Wb1;
    int Kpad = (layer == 0) ? KPAD0 : KPAD1;
    int Fext = (layer == 0) ? 32 : 512;
    for (int s = 0; s < L_STEPS; ++s) {
      const unsigned short* Aext = (layer == 0)
          ? a0   + (size_t)s * MROWS * 32
          : out0 + (size_t)s * MROWS * 512;
      dim3 grid(SIXH / 128, MROWS / 64);     // 12 x 24
      wmma_gate_gemm<<<grid, 256, 0, stream>>>(Xhid, Aext, Fext, Wb, Kpad, gate);
      float* hin  = (s & 1) ? hcolB : hcolA;
      float* hout = (s & 1) ? hcolA : hcolB;
      unsigned short* ob = (layer == 0) ? out0 + (size_t)s * MROWS * 512 : nullptr;
      gating_step<<<MROWS, 256, 0, stream>>>(
          gate, Bias + (size_t)layer * SIXH, hrow, hin, hout, Xhid, ob,
          hid_row, hid_col, s, layer);
    }
  }

  // --- decoder ---
  fc1_kernel<<<(BATCH * C_COLS * 512) / 256, 256, 0, stream>>>(
      hid_row, hid_col, fc1_W, fc1_b, last0);
  final_kernel<<<(BATCH * 96 * 21 + 255) / 256, 256, 0, stream>>>(
      last0, x_mark_dec, temb_W, fc2_W, fc2_b, x_enc, out);
}
